// QuantumResidualRegressor_78503412236645
// MI455X (gfx1250) — compile-verified, matrix operand order checked by reading, not run on hardware
//
#include <hip/hip_runtime.h>
#include <hip/hip_bf16.h>
#include <math.h>

typedef __attribute__((ext_vector_type(16))) _Float16 v16h;
typedef __attribute__((ext_vector_type(8)))  float    v8f;

#define N_QUBITS 6
#define N_LAYERS 3
#define D_IN  32
#define HID   16
#define ANG_STRIDE 8   // padded per-sample angle stride in workspace

// ---------------------------------------------------------------------------
// Branchless fast math: hw v_exp_f32 (__expf) + hw v_rcp_f32 (amdgcn_rcpf).
// No exec-mask divergence, no IEEE div_scale/div_fixup sequences.
// ---------------------------------------------------------------------------
__device__ __forceinline__ float rcp_fast(float x) {
    return __builtin_amdgcn_rcpf(x);       // single v_rcp_f32, ~1 ulp
}

__device__ __forceinline__ float erf_fast(float x) {
    // Abramowitz & Stegun 7.1.26, |err| < 1.5e-7, branchless.
    const float ax = fabsf(x);
    const float t  = rcp_fast(fmaf(0.3275911f, ax, 1.0f));
    float y = fmaf(t, 1.061405429f, -1.453152027f);
    y = fmaf(t, y, 1.421413741f);
    y = fmaf(t, y, -0.284496736f);
    y = fmaf(t, y, 0.254829592f);
    y = y * t;
    const float r = fmaf(-y, __expf(-ax * ax), 1.0f);
    return copysignf(r, x);
}

__device__ __forceinline__ float gelu_fast(float x) {
    return 0.5f * x * (1.0f + erf_fast(x * 0.70710678118654752f));
}

__device__ __forceinline__ float tanh_fast(float x) {
    // tanh(x) = 1 - 2/(exp(2x)+1); saturates correctly for |x| large.
    const float e = __expf(2.0f * x);
    return fmaf(-2.0f, rcp_fast(e + 1.0f), 1.0f);
}

// ---------------------------------------------------------------------------
// Kernel A: angles = tanh( gelu(X @ W1 + b1) @ W2 + b2 )   via two WMMAs/wave
// Each wave handles a 16-sample tile. 8 waves per 256-thread block.
// ---------------------------------------------------------------------------
__global__ void __launch_bounds__(256)
enc_wmma_kernel(const float* __restrict__ X,
                const float* __restrict__ W1, const float* __restrict__ B1,
                const float* __restrict__ W2, const float* __restrict__ B2,
                float* __restrict__ angles, int Bn) {
    __shared__ float lds[8][16][17];        // [wave][row][col], padded stride 17

    const int lane = threadIdx.x & 31;
    const int wave = threadIdx.x >> 5;
    const int tile = blockIdx.x * 8 + wave;
    const int row0 = tile * 16;

    const int m     = lane & 15;            // row (A) / col (B,C) index
    const int half  = lane >> 4;            // lane half selects K sub-range
    const int kbase = half * 8;

    // ---- A fragment: 16 samples x 32 features (f16), ISA 16-bit A layout ----
    int mrow = row0 + m; if (mrow >= Bn) mrow = Bn - 1;   // clamp for tails
    const float* xr = X + (size_t)mrow * D_IN;
    v16h a;
    #pragma unroll
    for (int j = 0; j < 8; ++j) a[j]     = (_Float16)xr[kbase + j];
    #pragma unroll
    for (int j = 0; j < 8; ++j) a[8 + j] = (_Float16)xr[kbase + 16 + j];

    // ---- B fragment: W1 column n per lane (mirror of A layout) ----
    const int n = m;
    v16h b;
    #pragma unroll
    for (int j = 0; j < 8; ++j) b[j]     = (_Float16)W1[(kbase + j) * HID + n];
    #pragma unroll
    for (int j = 0; j < 8; ++j) b[8 + j] = (_Float16)W1[(kbase + 16 + j) * HID + n];

    v8f c = {};
    c = __builtin_amdgcn_wmma_f32_16x16x32_f16(false, a, false, b, (short)0, c,
                                               false, false);

    // bias + GELU; stage H tile in LDS (C layout -> row-major)
    const float bias1 = B1[n];
    #pragma unroll
    for (int r = 0; r < 8; ++r) {
        lds[wave][half * 8 + r][n] = gelu_fast(c[r] + bias1);
    }
    __syncthreads();   // order LDS write->read (waves use disjoint slices)

    // ---- A2 fragment: H rows from LDS, K padded 16 -> 32 with zeros ----
    v16h a2;
    #pragma unroll
    for (int j = 0; j < 16; ++j) {
        const int k = (j < 8) ? (kbase + j) : (kbase + 8 + j);
        a2[j] = (k < 16) ? (_Float16)lds[wave][m][k] : (_Float16)0.0f;
    }

    // ---- B2 fragment: W2 [16 x 6], zero-padded to 32 x 16 ----
    v16h b2v;
    #pragma unroll
    for (int j = 0; j < 16; ++j) {
        const int k = (j < 8) ? (kbase + j) : (kbase + 8 + j);
        float w = 0.0f;
        if (k < 16 && n < N_QUBITS) w = W2[k * N_QUBITS + n];
        b2v[j] = (_Float16)w;
    }

    v8f c2 = {};
    c2 = __builtin_amdgcn_wmma_f32_16x16x32_f16(false, a2, false, b2v, (short)0,
                                                c2, false, false);

    if (n < N_QUBITS) {
        const float bias2 = B2[n];
        #pragma unroll
        for (int r = 0; r < 8; ++r) {
            const int row = row0 + half * 8 + r;
            if (row < Bn)
                angles[(size_t)row * ANG_STRIDE + n] = tanh_fast(c2[r] + bias2);
        }
    }
}

// ---------------------------------------------------------------------------
// Kernel B: 6-qubit statevector sim fully in registers + head MLP + residual.
// One thread per sample. Qubit q <-> bit (5-q) of the flattened index.
// ---------------------------------------------------------------------------
__device__ __forceinline__ void cnot_ring(float sr[64], float si[64]) {
    #pragma unroll
    for (int cq = 0; cq < N_QUBITS; ++cq) {
        const int tq = (cq + 1) % N_QUBITS;
        const int cm = 1 << (5 - cq);
        const int tm = 1 << (5 - tq);
        #pragma unroll
        for (int i = 0; i < 64; ++i) {
            if ((i & cm) && !(i & tm)) {
                const int j = i | tm;
                float tr = sr[i]; sr[i] = sr[j]; sr[j] = tr;
                float ti = si[i]; si[i] = si[j]; si[j] = ti;
            }
        }
    }
}

__device__ __forceinline__ void rx_all(float sr[64], float si[64],
                                       const float cw[N_QUBITS],
                                       const float sw[N_QUBITS]) {
    #pragma unroll
    for (int q = 0; q < N_QUBITS; ++q) {
        const int qm = 1 << (5 - q);
        const float c = cw[q], s = sw[q];
        #pragma unroll
        for (int i = 0; i < 64; ++i) {
            if (!(i & qm)) {
                const int j = i | qm;
                const float a0r = sr[i], a0i = si[i];
                const float a1r = sr[j], a1i = si[j];
                // RX = [[c, -i s], [-i s, c]]
                sr[i] = c * a0r + s * a1i;
                si[i] = c * a0i - s * a1r;
                sr[j] = c * a1r + s * a0i;
                si[j] = c * a1i - s * a0r;
            }
        }
    }
}

__global__ void __launch_bounds__(256)
qsim_head_kernel(const float* __restrict__ angles,
                 const float* __restrict__ qw,
                 const float* __restrict__ hw1, const float* __restrict__ hb1,
                 const float* __restrict__ hw2, const float* __restrict__ hb2,
                 const float* __restrict__ baseline,
                 float* __restrict__ out, int Bn) {
    const int bidx = blockIdx.x * blockDim.x + threadIdx.x;
    if (bidx >= Bn) return;
    const float* ar = angles + (size_t)bidx * ANG_STRIDE;

    // Per-qubit factor = RX(w_layer0) * RY(angle) applied to |0>:
    //   u0 = (cw*ca, -sw*sa), u1 = (cw*sa, -sw*ca)
    float f0r[N_QUBITS], f0i[N_QUBITS], f1r[N_QUBITS], f1i[N_QUBITS];
    #pragma unroll
    for (int q = 0; q < N_QUBITS; ++q) {
        const float ha = 0.5f * ar[q];
        const float ca = __cosf(ha), sa = __sinf(ha);
        const float hw = 0.5f * qw[q];               // layer 0 RX (uniform)
        const float cw = __cosf(hw), sw = __sinf(hw);
        f0r[q] = cw * ca; f0i[q] = -sw * sa;
        f1r[q] = cw * sa; f1i[q] = -sw * ca;
    }

    // Tensor-doubling build of the product state (RY embed + layer-0 RX).
    float sr[64], si[64];
    sr[0] = 1.0f; si[0] = 0.0f;
    #pragma unroll
    for (int q = 0; q < N_QUBITS; ++q) {
        const int sz = 1 << q;
        #pragma unroll
        for (int j = sz - 1; j >= 0; --j) {
            const float tr = sr[j], ti = si[j];
            sr[2 * j + 1] = tr * f1r[q] - ti * f1i[q];
            si[2 * j + 1] = tr * f1i[q] + ti * f1r[q];
            sr[2 * j]     = tr * f0r[q] - ti * f0i[q];
            si[2 * j]     = tr * f0i[q] + ti * f0r[q];
        }
    }
    cnot_ring(sr, si);                                // layer 0 entangler

    // Layers 1..2: full-state RX then CNOT ring. Weights are wave-uniform.
    #pragma unroll
    for (int l = 1; l < N_LAYERS; ++l) {
        float cw[N_QUBITS], sw[N_QUBITS];
        #pragma unroll
        for (int q = 0; q < N_QUBITS; ++q) {
            const float hw = 0.5f * qw[l * N_QUBITS + q];
            cw[q] = __cosf(hw); sw[q] = __sinf(hw);
        }
        rx_all(sr, si, cw, sw);
        cnot_ring(sr, si);
    }

    // PauliZ expvals via pairwise sum/diff tree.
    // Level j pairs over bit j: diff-sum = Z for qubit (5-j); pair-sums feed
    // the next level. 183 adds vs 384 for the naive signed sums.
    float p[64];
    #pragma unroll
    for (int i = 0; i < 64; ++i) p[i] = fmaf(sr[i], sr[i], si[i] * si[i]);
    float z[N_QUBITS];
    #pragma unroll
    for (int lev = 0; lev < 6; ++lev) {
        const int half_n = 64 >> (lev + 1);
        float acc = 0.0f;
        #pragma unroll
        for (int k = 0; k < half_n; ++k) {
            const float e = p[2 * k], o = p[2 * k + 1];
            acc += (e - o);
            p[k] = e + o;      // next-level pair sums (in place, k <= 2k)
        }
        z[5 - lev] = acc;
    }

    // Head MLP (6 -> 16 -> 1), GELU, plus residual baseline.
    float racc = hb2[0];
    #pragma unroll
    for (int o = 0; o < HID; ++o) {
        float h = hb1[o];
        #pragma unroll
        for (int q = 0; q < N_QUBITS; ++q) h = fmaf(z[q], hw1[q * HID + o], h);
        racc = fmaf(gelu_fast(h), hw2[o], racc);
    }
    out[bidx] = baseline[bidx] + racc;
}

// ---------------------------------------------------------------------------
extern "C" void kernel_launch(void* const* d_in, const int* in_sizes, int n_in,
                              void* d_out, int out_size, void* d_ws, size_t ws_size,
                              hipStream_t stream) {
    const float* features  = (const float*)d_in[0];
    const float* baseline  = (const float*)d_in[1];
    const float* enc_w1    = (const float*)d_in[2];
    const float* enc_b1    = (const float*)d_in[3];
    const float* enc_w2    = (const float*)d_in[4];
    const float* enc_b2    = (const float*)d_in[5];
    const float* q_weights = (const float*)d_in[6];
    const float* head_w1   = (const float*)d_in[7];
    const float* head_b1   = (const float*)d_in[8];
    const float* head_w2   = (const float*)d_in[9];
    const float* head_b2   = (const float*)d_in[10];
    float* out    = (float*)d_out;
    float* angles = (float*)d_ws;          // B * ANG_STRIDE floats

    const int Bn = in_sizes[0] / D_IN;

    const int blocksA = (Bn + 127) / 128;  // 8 waves * 16 samples per block
    enc_wmma_kernel<<<blocksA, 256, 0, stream>>>(features, enc_w1, enc_b1,
                                                 enc_w2, enc_b2, angles, Bn);

    const int blocksB = (Bn + 255) / 256;
    qsim_head_kernel<<<blocksB, 256, 0, stream>>>(angles, q_weights,
                                                  head_w1, head_b1,
                                                  head_w2, head_b2,
                                                  baseline, out, Bn);
}
